// TreeLSTMCell_12996571037754
// MI455X (gfx1250) — compile-verified, matrix operand order checked by reading, not run on hardware
//
#include <hip/hip_runtime.h>
#include <hip/hip_bf16.h>
#include <math.h>

// ---------------------------------------------------------------------------
// TreeLSTM cell for MI455X (gfx1250): bf16 WMMA GEMMs + fused segment-sums.
// ---------------------------------------------------------------------------

typedef __attribute__((ext_vector_type(16))) __bf16 v16bf;
typedef __attribute__((ext_vector_type(8)))  __bf16 v8bf;
typedef __attribute__((ext_vector_type(8)))  float  v8f;
typedef __attribute__((ext_vector_type(4)))  float  v4f;

#define HDIM   256
#define NNODES 32768
#define NEDGES 262144

#define BM 128          // block tile M
#define BN 64           // block tile N (per column-group)
#define BK 32           // K step == WMMA K
#define ASTR 40         // LDS row stride (bf16 elems), padded vs 32 for banks

__device__ __forceinline__ float sigmoidf_(float x) {
    return 1.0f / (1.0f + __expf(-x));
}

__device__ __forceinline__ v8f wmma_bf16(v16bf a, v16bf b, v8f c) {
    // (neg_a, A, neg_b, B, c_mod, C, reuse_a, reuse_b)
    return __builtin_amdgcn_wmma_f32_16x16x32_bf16(false, a, false, b,
                                                   (short)0, c, false, false);
}

// Build a 16x32 bf16 A/B fragment from an LDS tile (row-major, stride ASTR).
// ISA layout: lane l -> row (l&15); lanes 0-15 hold K=0..7,16..23;
// lanes 16-31 hold K=8..15,24..31.  Two 16B ds loads per lane.
__device__ __forceinline__ v16bf load_frag(const __bf16* lds, int row_base, int lane) {
    const __bf16* p = lds + (row_base + (lane & 15)) * ASTR + ((lane >> 4) * 8);
    v8bf lo = *(const v8bf*)(p);
    v8bf hi = *(const v8bf*)(p + 16);
    v16bf f;
#pragma unroll
    for (int i = 0; i < 8; ++i) { f[i] = lo[i]; f[8 + i] = hi[i]; }
    return f;
}

// Stage a BM x BK fp32 tile -> bf16 LDS tile. 256 threads, 16 floats each.
__device__ __forceinline__ void stage_A_f32(__bf16* sA, const float* __restrict__ src,
                                            int m0, int k0, int t,
                                            v4f* keep /*4 regs, raw fp32 copy*/) {
    int r = t >> 1, c = (t & 1) * 16;
    const float* g = src + (size_t)(m0 + r) * HDIM + k0 + c;
    v4f f0 = *(const v4f*)(g);
    v4f f1 = *(const v4f*)(g + 4);
    v4f f2 = *(const v4f*)(g + 8);
    v4f f3 = *(const v4f*)(g + 12);
    if (keep) { keep[0] = f0; keep[1] = f1; keep[2] = f2; keep[3] = f3; }
    v8bf h0, h1;
#pragma unroll
    for (int i = 0; i < 4; ++i) {
        h0[i] = (__bf16)f0[i]; h0[4 + i] = (__bf16)f1[i];
        h1[i] = (__bf16)f2[i]; h1[4 + i] = (__bf16)f3[i];
    }
    __bf16* d = sA + r * ASTR + c;
    *(v8bf*)(d)     = h0;
    *(v8bf*)(d + 8) = h1;
}

// Stage a BN x BK bf16 weight tile (row-major [rows][HDIM]) into LDS.
__device__ __forceinline__ void stage_B_bf16(__bf16* sB, const __bf16* __restrict__ w,
                                             int n0, int k0, int t) {
    int r = t >> 2, c = (t & 3) * 8;
    *(v8bf*)(sB + r * ASTR + c) =
        *(const v8bf*)(w + (size_t)(n0 + r) * HDIM + k0 + c);
}

// ---------------------------------------------------------------------------
__global__ void k_zero(float* __restrict__ p, int n) {
    int i = blockIdx.x * blockDim.x + threadIdx.x;
    if (i < n) p[i] = 0.0f;
}

__global__ void k_cvt_bf16(const float* __restrict__ s, __bf16* __restrict__ d, int n) {
    int i = blockIdx.x * blockDim.x + threadIdx.x;
    if (i < n) d[i] = (__bf16)s[i];
}

// ---------------------------------------------------------------------------
// GEMM 1: wf_x[m,n] = x[m,:] @ W_f[n,:] + W_f_b[n]        (M=32768, N=K=256)
__global__ __launch_bounds__(256) void k_gemm_wfx(const float* __restrict__ x,
                                                  const __bf16* __restrict__ Wf,
                                                  const float* __restrict__ bfb,
                                                  float* __restrict__ wf_x) {
    __shared__ __bf16 sA[BM * ASTR];
    __shared__ __bf16 sB[BN * ASTR];
    int m0 = blockIdx.x * BM, n0 = blockIdx.y * BN;
    int t = threadIdx.x, lane = t & 31, w = t >> 5;
    int wm = (w >> 1) * 32, wn = (w & 1) * 32;
    v8f acc[2][2] = {};

    for (int k0 = 0; k0 < HDIM; k0 += BK) {
        __syncthreads();
        stage_A_f32(sA, x, m0, k0, t, nullptr);
        stage_B_bf16(sB, Wf, n0, k0, t);
        __syncthreads();
        v16bf a0 = load_frag(sA, wm, lane);
        v16bf a1 = load_frag(sA, wm + 16, lane);
        v16bf b0 = load_frag(sB, wn, lane);
        v16bf b1 = load_frag(sB, wn + 16, lane);
        acc[0][0] = wmma_bf16(a0, b0, acc[0][0]);
        acc[0][1] = wmma_bf16(a0, b1, acc[0][1]);
        acc[1][0] = wmma_bf16(a1, b0, acc[1][0]);
        acc[1][1] = wmma_bf16(a1, b1, acc[1][1]);
    }
    int mrow = m0 + wm + 8 * (lane >> 4);
    int ncol = n0 + wn + (lane & 15);
#pragma unroll
    for (int tm = 0; tm < 2; ++tm)
#pragma unroll
        for (int tn = 0; tn < 2; ++tn) {
            int nn = ncol + tn * 16;
            float bb = bfb[nn];
#pragma unroll
            for (int v = 0; v < 8; ++v) {
                int mm = mrow + tm * 16 + v;
                wf_x[(size_t)mm * HDIM + nn] = acc[tm][tn][v] + bb;
            }
        }
}

// ---------------------------------------------------------------------------
// GEMM 2 (dominant): per-edge forget gate + fused segment sums.
//   f = sigmoid(child_h @ U_f.T + wf_x[parent]);  atomic c_agg += f*child_c
//   h_sum += child_h (fused into A-staging, n-block 0 only)
__global__ __launch_bounds__(256) void k_gemm_edges(const float* __restrict__ child_h,
                                                    const float* __restrict__ child_c,
                                                    const int* __restrict__ parent_idx,
                                                    const __bf16* __restrict__ Uf,
                                                    const float* __restrict__ wf_x,
                                                    float* __restrict__ c_agg,
                                                    float* __restrict__ h_sum) {
    __shared__ __bf16 sA[BM * ASTR];
    __shared__ __bf16 sB[BN * ASTR];
    int m0 = blockIdx.x * BM, n0 = blockIdx.y * BN;
    int t = threadIdx.x, lane = t & 31, w = t >> 5;
    int wm = (w >> 1) * 32, wn = (w & 1) * 32;
    v8f acc[2][2] = {};

    bool do_hsum = (blockIdx.y == 0);
    int my_row = m0 + (t >> 1);
    int my_col = (t & 1) * 16;
    int my_par = parent_idx[my_row];

    for (int k0 = 0; k0 < HDIM; k0 += BK) {
        __syncthreads();
        v4f raw[4];
        stage_A_f32(sA, child_h, m0, k0, t, raw);
        stage_B_bf16(sB, Uf, n0, k0, t);
        if (do_hsum) {
            float* hs = h_sum + (size_t)my_par * HDIM + k0 + my_col;
#pragma unroll
            for (int q = 0; q < 4; ++q)
#pragma unroll
                for (int j = 0; j < 4; ++j)
                    atomicAdd(hs + q * 4 + j, raw[q][j]);
        }
        __syncthreads();
        v16bf a0 = load_frag(sA, wm, lane);
        v16bf a1 = load_frag(sA, wm + 16, lane);
        v16bf b0 = load_frag(sB, wn, lane);
        v16bf b1 = load_frag(sB, wn + 16, lane);
        acc[0][0] = wmma_bf16(a0, b0, acc[0][0]);
        acc[0][1] = wmma_bf16(a0, b1, acc[0][1]);
        acc[1][0] = wmma_bf16(a1, b0, acc[1][0]);
        acc[1][1] = wmma_bf16(a1, b1, acc[1][1]);
    }
    int mrow = m0 + wm + 8 * (lane >> 4);
    int ncol = n0 + wn + (lane & 15);
#pragma unroll
    for (int tm = 0; tm < 2; ++tm)
#pragma unroll
        for (int tn = 0; tn < 2; ++tn) {
            int nn = ncol + tn * 16;
#pragma unroll
            for (int v = 0; v < 8; ++v) {
                int mm = mrow + tm * 16 + v;
                int p = parent_idx[mm];
                float fg = sigmoidf_(acc[tm][tn][v] + wf_x[(size_t)p * HDIM + nn]);
                float cc = child_c[(size_t)mm * HDIM + nn];
                atomicAdd(&c_agg[(size_t)p * HDIM + nn], fg * cc);
            }
        }
}

// ---------------------------------------------------------------------------
// GEMM 3: iou = h_sum @ U_iou.T + b_iou (3 column groups), fused gate epilogue.
__global__ __launch_bounds__(256) void k_gemm_iou(const float* __restrict__ h_sum,
                                                  const __bf16* __restrict__ Uiou,
                                                  const float* __restrict__ b_iou,
                                                  const float* __restrict__ c_agg,
                                                  float* __restrict__ out_h,
                                                  float* __restrict__ out_c) {
    __shared__ __bf16 sA[BM * ASTR];
    __shared__ __bf16 sB[3 * BN * ASTR];
    int m0 = blockIdx.x * BM, n0 = blockIdx.y * BN;
    int t = threadIdx.x, lane = t & 31, w = t >> 5;
    int wm = (w >> 1) * 32, wn = (w & 1) * 32;
    v8f acc[3][2][2] = {};

    for (int k0 = 0; k0 < HDIM; k0 += BK) {
        __syncthreads();
        stage_A_f32(sA, h_sum, m0, k0, t, nullptr);
#pragma unroll
        for (int g = 0; g < 3; ++g)
            stage_B_bf16(sB + g * BN * ASTR, Uiou, g * HDIM + n0, k0, t);
        __syncthreads();
        v16bf a0 = load_frag(sA, wm, lane);
        v16bf a1 = load_frag(sA, wm + 16, lane);
#pragma unroll
        for (int g = 0; g < 3; ++g) {
            const __bf16* sg = sB + g * BN * ASTR;
            v16bf b0 = load_frag(sg, wn, lane);
            v16bf b1 = load_frag(sg, wn + 16, lane);
            acc[g][0][0] = wmma_bf16(a0, b0, acc[g][0][0]);
            acc[g][0][1] = wmma_bf16(a0, b1, acc[g][0][1]);
            acc[g][1][0] = wmma_bf16(a1, b0, acc[g][1][0]);
            acc[g][1][1] = wmma_bf16(a1, b1, acc[g][1][1]);
        }
    }
    int mrow = m0 + wm + 8 * (lane >> 4);
    int ncol = n0 + wn + (lane & 15);
#pragma unroll
    for (int tm = 0; tm < 2; ++tm)
#pragma unroll
        for (int tn = 0; tn < 2; ++tn) {
            int nn = ncol + tn * 16;
            float bi = b_iou[nn];
            float bo = b_iou[HDIM + nn];
            float bu = b_iou[2 * HDIM + nn];
#pragma unroll
            for (int v = 0; v < 8; ++v) {
                int mm = mrow + tm * 16 + v;
                float iv = acc[0][tm][tn][v] + bi;
                float ov = acc[1][tm][tn][v] + bo;
                float uv = acc[2][tm][tn][v] + bu;
                float ca = c_agg[(size_t)mm * HDIM + nn];
                float c  = sigmoidf_(iv) * tanhf(uv) + ca;
                float h  = sigmoidf_(ov) * tanhf(c);
                out_h[(size_t)mm * HDIM + nn] = h;
                out_c[(size_t)mm * HDIM + nn] = c;
            }
        }
}

// ---------------------------------------------------------------------------
extern "C" void kernel_launch(void* const* d_in, const int* in_sizes, int n_in,
                              void* d_out, int out_size, void* d_ws, size_t ws_size,
                              hipStream_t stream) {
    const float* child_h    = (const float*)d_in[0];
    const float* child_c    = (const float*)d_in[1];
    const float* x          = (const float*)d_in[2];
    const int*   parent_idx = (const int*)  d_in[3];
    const float* U_iou_w    = (const float*)d_in[4];
    const float* b_iou      = (const float*)d_in[5];
    const float* U_f_w      = (const float*)d_in[6];
    const float* W_f_w      = (const float*)d_in[7];
    const float* W_f_b      = (const float*)d_in[8];

    float*  wf_x   = (float*)d_ws;                       // [N*H] fp32
    float*  c_agg  = wf_x  + (size_t)NNODES * HDIM;      // [N*H] fp32
    float*  h_sum  = c_agg + (size_t)NNODES * HDIM;      // [N*H] fp32
    __bf16* Wf_bf  = (__bf16*)(h_sum + (size_t)NNODES * HDIM);
    __bf16* Uf_bf  = Wf_bf + HDIM * HDIM;
    __bf16* Ui_bf  = Uf_bf + HDIM * HDIM;

    float* out_h = (float*)d_out;
    float* out_c = out_h + (size_t)NNODES * HDIM;

    // zero the atomic accumulators (c_agg, h_sum are contiguous)
    {
        int n = 2 * NNODES * HDIM;
        k_zero<<<(n + 255) / 256, 256, 0, stream>>>(c_agg, n);
    }
    // weights -> bf16
    k_cvt_bf16<<<(HDIM * HDIM + 255) / 256, 256, 0, stream>>>(W_f_w, Wf_bf, HDIM * HDIM);
    k_cvt_bf16<<<(HDIM * HDIM + 255) / 256, 256, 0, stream>>>(U_f_w, Uf_bf, HDIM * HDIM);
    k_cvt_bf16<<<(3 * HDIM * HDIM + 255) / 256, 256, 0, stream>>>(U_iou_w, Ui_bf, 3 * HDIM * HDIM);

    // GEMM1: wf_x
    k_gemm_wfx<<<dim3(NNODES / BM, HDIM / BN), 256, 0, stream>>>(x, Wf_bf, W_f_b, wf_x);
    // GEMM2: edges (forget gates + fused segment sums)
    k_gemm_edges<<<dim3(NEDGES / BM, HDIM / BN), 256, 0, stream>>>(
        child_h, child_c, parent_idx, Uf_bf, wf_x, c_agg, h_sum);
    // GEMM3: iou + gate epilogue -> (h, c)
    k_gemm_iou<<<dim3(NNODES / BM, HDIM / BN), 256, 0, stream>>>(
        h_sum, Ui_bf, b_iou, c_agg, out_h, out_c);
}
